// ExpertMLP_37589553774754
// MI455X (gfx1250) — matrix operand order repaired and checked by reading, not compiled, not run on hardware
//
#include <hip/hip_runtime.h>
#include <math.h>

#define S_TOK 8192
#define H_DIM 1024
#define F_DIM 4096
#define E_EXP 8
#define K_TOP 2
#define NSLOT (S_TOK * K_TOP)

#define LSTR 40   // LDS row stride in bf16 units (32 data + 8 pad); 80B, 16B-aligned, conflict-free
#define TILE_ELEMS (128 * LSTR)

#if defined(__has_builtin)
#if __has_builtin(__builtin_amdgcn_global_load_async_to_lds_b128) && \
    __has_builtin(__builtin_amdgcn_s_wait_asynccnt)
#define HAVE_ASYNC_LDS 1
#endif
#endif

typedef __bf16 bf16x16 __attribute__((ext_vector_type(16)));
typedef float f32x8 __attribute__((ext_vector_type(8)));
typedef int v4i __attribute__((ext_vector_type(4)));
typedef __attribute__((address_space(1))) v4i* gv4i_p;   // global (AS1) pointer to int4
typedef __attribute__((address_space(3))) v4i* lv4i_p;   // LDS (AS3) pointer to int4

__device__ __forceinline__ unsigned int f2bf_raw(float f) {
    // round-to-nearest-even fp32 -> bf16 (raw 16-bit pattern in low bits)
    unsigned int u = __builtin_bit_cast(unsigned int, f);
    unsigned int r = u + 0x7FFFu + ((u >> 16) & 1u);
    return r >> 16;
}

// A fragment (16x32 bf16): lanes 0-15 hold M=lane, K=0..7 (v0-3) and K=16..23 (v4-7);
// lanes 16-31 hold K=8..15 and K=24..31.
__device__ __forceinline__ bf16x16 lds_fragA(const unsigned short* tile, int row0, int lane) {
    int r = row0 + (lane & 15);
    int kh = (lane >> 4) << 3; // 0 or 8
    union { uint4 u[2]; bf16x16 v; } U;
    U.u[0] = *reinterpret_cast<const uint4*>(tile + r * LSTR + kh);
    U.u[1] = *reinterpret_cast<const uint4*>(tile + r * LSTR + kh + 16);
    return U.v;
}

// B fragment (32x16 bf16 stored as B^T rows): lanes 0-15 hold N=lane, K=0..7 and K=16..23; lanes 16-31 K=8..15 and K=24..31.
__device__ __forceinline__ bf16x16 lds_fragB(const unsigned short* tile, int col0, int lane) {
    int n = col0 + (lane & 15);
    int kb = (lane >> 4) << 3; // 0 or 8
    union { uint4 u[2]; bf16x16 v; } U;
    U.u[0] = *reinterpret_cast<const uint4*>(tile + n * LSTR + kb);
    U.u[1] = *reinterpret_cast<const uint4*>(tile + n * LSTR + kb + 16);
    return U.v;
}

// ---------------- fp32 -> bf16 pre-conversion (bandwidth-bound, runs once) ----------------
__global__ void k_cvt_bf16(const float* __restrict__ src, unsigned short* __restrict__ dst, int n4) {
    int i = blockIdx.x * blockDim.x + threadIdx.x;
    if (i < n4) {
        float4 v = reinterpret_cast<const float4*>(src)[i];
        uint2 pk;
        pk.x = f2bf_raw(v.x) | (f2bf_raw(v.y) << 16);
        pk.y = f2bf_raw(v.z) | (f2bf_raw(v.w) << 16);
        reinterpret_cast<uint2*>(dst)[i] = pk;
    }
}

// ---------------- routing ----------------
__global__ void k_zero(int* cnt, int* cursor) {
    int t = threadIdx.x;
    if (t < E_EXP) { cnt[t] = 0; cursor[t] = 0; }
}

__global__ void k_count(const int* __restrict__ topk_e, int* cnt) {
    int i = blockIdx.x * blockDim.x + threadIdx.x;
    if (i < NSLOT) atomicAdd(&cnt[topk_e[i]], 1);
}

__global__ void k_scan(const int* __restrict__ cnt, int* off) {
    if (threadIdx.x == 0) {
        int a = 0;
        for (int e = 0; e < E_EXP; ++e) { off[e] = a; a += cnt[e]; }
        off[E_EXP] = a;
    }
}

__global__ void k_fill(const int* __restrict__ topk_e, const float* __restrict__ topk_w,
                       const int* __restrict__ off, int* cursor,
                       int* rowtok, float* roww, int* slot_of) {
    int i = blockIdx.x * blockDim.x + threadIdx.x;
    if (i < NSLOT) {
        int e = topk_e[i];
        int p = atomicAdd(&cursor[e], 1);
        int g = off[e] + p;
        rowtok[g] = i / K_TOP;
        roww[g] = topk_w[i];
        slot_of[i] = g;
    }
}

// ---------------- grouped bf16 GEMM (async-LDS double-buffered pipeline) ----------------
// PASS 1: hw[g, f] = bf16( silu( xbf[tok(g)] . W1bf[e]^T ) * gate(g) ),  ND=F, KD=H
// PASS 2: yp[g, h] = f32 ( hw[g] . W2bf[e]^T ),                           ND=H, KD=F
template <int PASS, int ND, int KD>
__global__ __launch_bounds__(256, 1) void moe_gemm(
    const unsigned short* __restrict__ Asrc, // bf16: xbf (pass1, gathered by token) or hw (pass2, by slot)
    const unsigned short* __restrict__ wBb,  // bf16 weights [E, ND, KD]
    unsigned short* __restrict__ hOut,       // pass1 output (bf16)
    float* __restrict__ ypOut,               // pass2 output (f32)
    const int* __restrict__ rowtok,
    const float* __restrict__ roww,
    const int* __restrict__ off,
    const int* __restrict__ cnt)
{
    const int e = blockIdx.z;
    const int cn = cnt[e];
    const int mt = blockIdx.y;
    if (mt * 128 >= cn) return;            // early-exit unused M tiles of this expert
    const int off_e = off[e];
    const int n0 = blockIdx.x * 128;
    const int tid = threadIdx.x;
    const int lane = tid & 31;
    const int w = tid >> 5;
    const int wm = (w & 3) * 32;           // 4 waves along M (128 rows)
    const int wn = (w >> 2) * 64;          // 2 waves along N (128 cols)

    __shared__ unsigned short As[2][TILE_ELEMS];
    __shared__ unsigned short Bs[2][TILE_ELEMS];

    f32x8 acc[2][4] = {};

    const unsigned short* wBe = wBb + (size_t)e * ND * KD;

    // --- K-invariant per-thread staging indices (gather resolved once) ---
    // Tile = 128 rows x 32 k of bf16 -> 512 uint4 chunks -> 2 per thread.
    const unsigned short* gpA[2];
    const unsigned short* gpB[2];
    int lsA[2], lsB[2];
#pragma unroll
    for (int j = 0; j < 2; ++j) {
        int i = tid + j * 256;
        int r = i >> 2, c8 = (i & 3) << 3;
        int lr = mt * 128 + r;
        int g = off_e + (lr < cn ? lr : cn - 1);   // clamp padding rows
        size_t row = (PASS == 1) ? (size_t)rowtok[g] : (size_t)g;
        gpA[j] = Asrc + row * KD + c8;
        lsA[j] = r * LSTR + c8;
        gpB[j] = wBe + (size_t)(n0 + r) * KD + c8;
        lsB[j] = r * LSTR + c8;
    }

    // One K-step of WMMA from LDS buffer `buf`.  B fragments are loaded per-ni and
    // consumed immediately (short live ranges -> low register pressure).
    auto compute = [&](int buf) {
        bf16x16 afr0 = lds_fragA(As[buf], wm, lane);
        bf16x16 afr1 = lds_fragA(As[buf], wm + 16, lane);
#pragma unroll
        for (int ni = 0; ni < 4; ++ni) {
            bf16x16 bfr = lds_fragB(Bs[buf], wn + ni * 16, lane);
            acc[0][ni] = __builtin_amdgcn_wmma_f32_16x16x32_bf16(
                false, afr0, false, bfr, (short)0, acc[0][ni], false, false);
            acc[1][ni] = __builtin_amdgcn_wmma_f32_16x16x32_bf16(
                false, afr1, false, bfr, (short)0, acc[1][ni], false, false);
        }
    };

#if defined(HAVE_ASYNC_LDS)
    // ---- async global->LDS staging: no staging VGPRs, DMA overlaps WMMA ----
    auto stage = [&](int buf, int k0) {
#pragma unroll
        for (int j = 0; j < 2; ++j) {
            __builtin_amdgcn_global_load_async_to_lds_b128(
                (gv4i_p)(gpA[j] + k0), (lv4i_p)&As[buf][lsA[j]], 0, 0);
            __builtin_amdgcn_global_load_async_to_lds_b128(
                (gv4i_p)(gpB[j] + k0), (lv4i_p)&Bs[buf][lsB[j]], 0, 0);
        }
    };

    stage(0, 0);
    __builtin_amdgcn_s_wait_asynccnt(0);
    __syncthreads();

    stage(1, 32);            // step 1 -> buffer 1 (DMA runs during compute below)
    compute(0);
    __builtin_amdgcn_s_wait_asynccnt(0);
    __syncthreads();

#pragma clang loop unroll(disable)
    for (int k0 = 64; k0 <= KD - 64; k0 += 64) {
        stage(0, k0);        // even step -> buffer 0
        compute(1);
        __builtin_amdgcn_s_wait_asynccnt(0);
        __syncthreads();

        stage(1, k0 + 32);   // odd step -> buffer 1
        compute(0);
        __builtin_amdgcn_s_wait_asynccnt(0);
        __syncthreads();
    }
    compute(1);              // final (odd) staged step
#else
    // ---- fallback: register-staged copies ----
    uint4 av[2], bv[2];
    auto gload = [&](int k0) {
#pragma unroll
        for (int j = 0; j < 2; ++j) av[j] = *reinterpret_cast<const uint4*>(gpA[j] + k0);
#pragma unroll
        for (int j = 0; j < 2; ++j) bv[j] = *reinterpret_cast<const uint4*>(gpB[j] + k0);
    };
    auto lstore = [&](int buf) {
#pragma unroll
        for (int j = 0; j < 2; ++j) *reinterpret_cast<uint4*>(&As[buf][lsA[j]]) = av[j];
#pragma unroll
        for (int j = 0; j < 2; ++j) *reinterpret_cast<uint4*>(&Bs[buf][lsB[j]]) = bv[j];
    };

    gload(0);
    lstore(0);
    __syncthreads();

    gload(32);
    compute(0);
    lstore(1);
    __syncthreads();

#pragma clang loop unroll(disable)
    for (int k0 = 64; k0 <= KD - 64; k0 += 64) {
        gload(k0);
        compute(1);
        lstore(0);
        __syncthreads();

        gload(k0 + 32);
        compute(0);
        lstore(1);
        __syncthreads();
    }
    compute(1);
#endif

    // ---- epilogue: C/D layout: VGPR j, lanes 0-15 -> M=j, lanes 16-31 -> M=8+j; N=lane&15 ----
#pragma unroll
    for (int mi = 0; mi < 2; ++mi) {
        int lrbase = wm + mi * 16 + ((lane >> 4) << 3);
#pragma unroll
        for (int j = 0; j < 8; ++j) {
            int lr = mt * 128 + lrbase + j;
            if (lr < cn) {
                int g = off_e + lr;
                float gate = (PASS == 1) ? roww[g] : 0.0f;
#pragma unroll
                for (int ni = 0; ni < 4; ++ni) {
                    int col = n0 + wn + ni * 16 + (lane & 15);
                    float v = acc[mi][ni][j];
                    if (PASS == 1) {
                        // silu via fast reciprocal: v * rcp(1 + e^-v)
                        float sv = v * __builtin_amdgcn_rcpf(1.0f + __expf(-v)) * gate;
                        hOut[(size_t)g * ND + col] = (unsigned short)f2bf_raw(sv);
                    } else {
                        ypOut[(size_t)g * ND + col] = v;
                    }
                }
            }
        }
    }
}

// ---------------- combine ----------------
__global__ void k_combine(const float* __restrict__ yp, const int* __restrict__ slot_of,
                          float* __restrict__ y) {
    int idx = blockIdx.x * blockDim.x + threadIdx.x;
    if (idx < S_TOK * H_DIM) {
        int s = idx / H_DIM;
        int h = idx - s * H_DIM;
        float a = yp[(size_t)slot_of[s * K_TOP + 0] * H_DIM + h];
        float b = yp[(size_t)slot_of[s * K_TOP + 1] * H_DIM + h];
        y[idx] = a + b;
    }
}

extern "C" void kernel_launch(void* const* d_in, const int* in_sizes, int n_in,
                              void* d_out, int out_size, void* d_ws, size_t ws_size,
                              hipStream_t stream) {
    (void)in_sizes; (void)n_in; (void)out_size; (void)ws_size;
    const float* x      = (const float*)d_in[0];
    const int*   topk_e = (const int*)d_in[1];
    const float* topk_w = (const float*)d_in[2];
    const float* w1     = (const float*)d_in[3];
    const float* w2     = (const float*)d_in[4];
    float* y = (float*)d_out;

    char* p = (char*)d_ws;
    auto alloc = [&](size_t bytes, size_t align) -> char* {
        size_t a = (size_t)p;
        a = (a + align - 1) & ~(align - 1);
        p = (char*)a;
        char* r = p;
        p += bytes;
        return r;
    };
    int*   cnt     = (int*)alloc(E_EXP * 4, 16);
    int*   cursor  = (int*)alloc(E_EXP * 4, 16);
    int*   off     = (int*)alloc((E_EXP + 1) * 4, 16);
    int*   rowtok  = (int*)alloc((size_t)NSLOT * 4, 16);
    float* roww    = (float*)alloc((size_t)NSLOT * 4, 16);
    int*   slot_of = (int*)alloc((size_t)NSLOT * 4, 16);
    unsigned short* xbf  = (unsigned short*)alloc((size_t)S_TOK * H_DIM * 2, 256);         // 17 MB
    unsigned short* w1bf = (unsigned short*)alloc((size_t)E_EXP * F_DIM * H_DIM * 2, 256); // 67 MB
    unsigned short* w2bf = (unsigned short*)alloc((size_t)E_EXP * H_DIM * F_DIM * 2, 256); // 67 MB
    unsigned short* hw   = (unsigned short*)alloc((size_t)NSLOT * F_DIM * 2, 256);         // 134 MB
    float* yp            = (float*)alloc((size_t)NSLOT * H_DIM * 4, 256);                  // 67 MB

    // pre-convert activations + weights to bf16 (one-shot, bandwidth-bound)
    {
        int n4x = (S_TOK * H_DIM) / 4;
        int n4w = (E_EXP * F_DIM * H_DIM) / 4;
        k_cvt_bf16<<<(n4x + 255) / 256, 256, 0, stream>>>(x, xbf, n4x);
        k_cvt_bf16<<<(n4w + 255) / 256, 256, 0, stream>>>(w1, w1bf, n4w);
        k_cvt_bf16<<<(n4w + 255) / 256, 256, 0, stream>>>(w2, w2bf, n4w);
    }

    k_zero<<<1, 64, 0, stream>>>(cnt, cursor);
    k_count<<<(NSLOT + 255) / 256, 256, 0, stream>>>(topk_e, cnt);
    k_scan<<<1, 1, 0, stream>>>(cnt, off);
    k_fill<<<(NSLOT + 255) / 256, 256, 0, stream>>>(topk_e, topk_w, off, cursor,
                                                    rowtok, roww, slot_of);

    dim3 g1(F_DIM / 128, NSLOT / 128, E_EXP);
    moe_gemm<1, F_DIM, H_DIM><<<g1, 256, 0, stream>>>(xbf, w1bf, hw, nullptr,
                                                      rowtok, roww, off, cnt);
    dim3 g2(H_DIM / 128, NSLOT / 128, E_EXP);
    moe_gemm<2, H_DIM, F_DIM><<<g2, 256, 0, stream>>>(hw, w2bf, nullptr, yp,
                                                      rowtok, roww, off, cnt);

    k_combine<<<(S_TOK * H_DIM) / 256, 256, 0, stream>>>(yp, slot_of, y);
}
